// LossYOLOv3_88167088652560
// MI455X (gfx1250) — compile-verified, matrix operand order checked by reading, not run on hardware
//
#include <hip/hip_runtime.h>
#include <math.h>

#define NT 10647
#define BIMG 32
#define NBOX 50
#define NCLS 20
#define NNEG 1000
#define EPSF 1e-7f

typedef float v2f __attribute__((ext_vector_type(2)));
typedef float v8f __attribute__((ext_vector_type(8)));
typedef _Float16 v16h __attribute__((ext_vector_type(16)));

__device__ __constant__ float c_anc[9][2] = {
    {10.f/416.f, 13.f/416.f},  {16.f/416.f, 30.f/416.f},  {33.f/416.f, 23.f/416.f},
    {30.f/416.f, 61.f/416.f},  {62.f/416.f, 45.f/416.f},  {59.f/416.f, 119.f/416.f},
    {116.f/416.f, 90.f/416.f}, {156.f/416.f, 198.f/416.f},{373.f/416.f, 326.f/416.f}};

// CIoU with the two atan() terms already computed (they are per-operand
// invariants: atan(w/(h+eps)) of each box), hoisted out of O(N*M) loops.
__device__ __forceinline__ float ciou_pre(float al, float at, float ar, float ab,
                                          float atanA,
                                          float bl, float bt, float br, float bb,
                                          float atanB) {
  float ltx = fmaxf(al, bl), lty = fmaxf(at, bt);
  float rbx = fminf(ar, br), rby = fminf(ab, bb);
  float iw = fmaxf(rbx - ltx, 0.0f), ih = fmaxf(rby - lty, 0.0f);
  float inter = iw * ih;
  float wa = ar - al, ha = ab - at;
  float wb = br - bl, hb = bb - bt;
  float uni = wa * ha + wb * hb - inter;
  float iou = inter / (uni + EPSF);
  float cw = fmaxf(ar, br) - fminf(al, bl);
  float ch = fmaxf(ab, bb) - fminf(at, bt);
  float c2 = cw * cw + ch * ch + EPSF;
  float dx = (al + ar) * 0.5f - (bl + br) * 0.5f;
  float dy = (at + ab) * 0.5f - (bt + bb) * 0.5f;
  float rho2 = dx * dx + dy * dy;
  float da = atanA - atanB;
  float v = 0.4052847345693511f * da * da;  // 4/pi^2
  float alpha = v / (1.0f - iou + v + EPSF);
  return iou - rho2 / c2 - alpha * v;
}

__device__ __forceinline__ float ciou_f(float al, float at, float ar, float ab,
                                        float bl, float bt, float br, float bb) {
  float atanA = atanf((ar - al) / (ab - at + EPSF));
  float atanB = atanf((br - bl) / (bb - bt + EPSF));
  return ciou_pre(al, at, ar, ab, atanA, bl, bt, br, bb, atanB);
}

__device__ __forceinline__ void anchor_decode(int a, int& lvl, int& f, int& cell, int& sub) {
  int r;
  if (a < 8112)       { lvl = 0; f = 52; r = a; }
  else if (a < 10140) { lvl = 1; f = 26; r = a - 8112; }
  else                { lvl = 2; f = 13; r = a - 10140; }
  cell = r / 3;
  sub = r - cell * 3;
}

// ------ Kernel 1: predicted LTRB + hoisted atan for every anchor ------------
__global__ void k_pred_ltrb(const float* __restrict__ p, float4* __restrict__ ltrb,
                            float* __restrict__ predAt) {
  int tid = blockIdx.x * blockDim.x + threadIdx.x;
  if (tid >= BIMG * NT) return;
  int a = tid % NT;
  int lvl, f, cell, sub;
  anchor_decode(a, lvl, f, cell, sub);
  const float* pp = p + (size_t)tid * 25;
  __builtin_prefetch(pp, 0, 0);
  float tx = pp[0], ty = pp[1], tw = pp[2], th = pp[3];
  float ff = (float)f;
  float ax = (float)(cell % f) / ff;
  float ay = (float)(cell / f) / ff;
  float bx = tx / ff + ax;
  float by = ty / ff + ay;
  float bw = expf(tw) * c_anc[lvl * 3 + sub][0];
  float bh = expf(th) * c_anc[lvl * 3 + sub][1];
  ltrb[tid] = make_float4(bx - bw * 0.5f, by - bh * 0.5f, bx + bw * 0.5f, by + bh * 0.5f);
  predAt[tid] = atanf(bw / (bh + EPSF));
}

// ---------------- Kernel 2: positive losses (cls + conf_p + box) ------------
__global__ void k_pos(const float* __restrict__ p, const float* __restrict__ boxes,
                      const int* __restrict__ labels, const float4* __restrict__ ltrb,
                      const float* __restrict__ predAt, float* __restrict__ posAcc) {
  int b = blockIdx.x;
  int i = threadIdx.x;  // 64 threads, first 50 active
  __shared__ float red[64];
  float partial = 0.0f;
  if (i < NBOX) {
    const float* bp = boxes + ((size_t)b * NBOX + i) * 4;
    float l = bp[0], t = bp[1], r = bp[2], btm = bp[3];
    float bcx = (l + r) * 0.5f, bcy = (t + btm) * 0.5f;
    float bw = r - l, bh = btm - t;
    float atanA = atanf(bw / (bh + EPSF));
    // best of 9 candidate anchors (CIoU is translation invariant -> the
    // reference's +ids offset trick reduces to a per-box 9-way argmax)
    float best = -INFINITY; int bk = 0;
    for (int k = 0; k < 9; ++k) {
      int lvl = k / 3;
      float f = (lvl == 0) ? 52.f : (lvl == 1 ? 26.f : 13.f);
      float ax = floorf(bcx * f) / f;
      float ay = floorf(bcy * f) / f;
      float aw = c_anc[k][0], ah = c_anc[k][1];
      float atanB = atanf(aw / (ah + EPSF));
      float v = ciou_pre(l, t, r, btm, atanA,
                         ax - aw * 0.5f, ay - ah * 0.5f, ax + aw * 0.5f, ay + ah * 0.5f,
                         atanB);
      if (v > best) { best = v; bk = k; }
    }
    int lvl = bk / 3;
    int fi = (lvl == 0) ? 52 : (lvl == 1 ? 26 : 13);
    int col = (int)floorf(bcx * (float)fi);
    int row = (int)floorf(bcy * (float)fi);
    int offc = (lvl == 0) ? 0 : (lvl == 1 ? 2704 : 3380);
    int midx = (offc + row * fi + col) * 3 + lvl;
    const float* pp = p + ((size_t)b * NT + midx) * 25;
    __builtin_prefetch(pp, 0, 0);
    int li = labels[b * NBOX + i] - 1;
    float lcls = 0.f;
    for (int c = 0; c < NCLS; ++c) {
      float pc = fminf(fmaxf(pp[5 + c], EPSF), 1.0f - EPSF);
      lcls += (c == li) ? -logf(pc) : -logf(1.0f - pc);
    }
    float pconf = fminf(fmaxf(pp[4], EPSF), 1.0f - EPSF);
    float lcp = -logf(pconf);
    float4 pl = ltrb[(size_t)b * NT + midx];
    float cp = ciou_pre(l, t, r, btm, atanA, pl.x, pl.y, pl.z, pl.w,
                        predAt[(size_t)b * NT + midx]);
    partial = lcls + lcp + (2.0f - bw * bh) * (1.0f - cp);
  }
  red[i] = partial;
  __syncthreads();
  for (int s = 32; s > 0; s >>= 1) {
    if (i < s) red[i] += red[i + s];
    __syncthreads();
  }
  if (i == 0) posAcc[b] = red[0];
}

// ---------------- Kernel 3: mx[b][a] = max_box CIoU(box, pred) --------------
// Hot loop: 32*10647*50 CIoUs; atans staged/precomputed, inner loop is pure
// min/max/FMA/div VALU work.
__global__ void k_mx(const float4* __restrict__ ltrb, const float* __restrict__ predAt,
                     const float* __restrict__ boxes, float* __restrict__ mx) {
  int b = blockIdx.y;
  int a = blockIdx.x * blockDim.x + threadIdx.x;
  __shared__ float4 sbox[NBOX];
  __shared__ float sat[NBOX];
  if (threadIdx.x < NBOX) {
    const float* bp = boxes + ((size_t)b * NBOX + threadIdx.x) * 4;
    float4 bx = make_float4(bp[0], bp[1], bp[2], bp[3]);
    sbox[threadIdx.x] = bx;
    sat[threadIdx.x] = atanf((bx.z - bx.x) / (bx.w - bx.y + EPSF));
  }
  __syncthreads();
  if (a >= NT) return;
  float4 pl = ltrb[(size_t)b * NT + a];
  float pat = predAt[(size_t)b * NT + a];
  float best = -INFINITY;
  #pragma unroll 5
  for (int i = 0; i < NBOX; ++i) {
    float4 bx = sbox[i];
    best = fmaxf(best, ciou_pre(bx.x, bx.y, bx.z, bx.w, sat[i],
                                pl.x, pl.y, pl.z, pl.w, pat));
  }
  mx[(size_t)b * NT + a] = best;
}

// ---------------- Kernel 4: hard-negative mining + conf_n loss --------------
// Replicates: key = mask ? mx : inf ; order = stable argsort(key);
//             rank = cumsum(mask)-1 ; neg_idx = rank[order][:1000];
//             sum(-log(clip(1 - p[neg_idx,4])))
#define NEG_TPB 1024
#define CPT 11  // ceil(10647/1024)
__global__ void __launch_bounds__(NEG_TPB)
k_neg(const float* __restrict__ p, const float* __restrict__ mx, float* __restrict__ negAcc) {
  int b = blockIdx.x;
  int t = threadIdx.x;
  __shared__ int sred[NEG_TPB];
  __shared__ int sred2[NEG_TPB];
  __shared__ float sfred[NEG_TPB];

  unsigned u[CPT];
  unsigned char m[CPT];
  const float* mxb = mx + (size_t)b * NT;
  int base = t * CPT;
  __builtin_prefetch(mxb + base, 0, 0);
  for (int j = 0; j < CPT; ++j) {
    int g = base + j;
    if (g < NT) {
      float v = mxb[g];
      int msk = v < 0.5f;
      m[j] = (unsigned char)msk;
      float key = msk ? v : INFINITY;
      unsigned bits = __float_as_uint(key);
      u[j] = (bits & 0x80000000u) ? ~bits : (bits | 0x80000000u);  // sortable uint
    } else { m[j] = 0; u[j] = 0xFFFFFFFFu; }
  }

  // Radix-select the 1000th smallest key. On exit: T = its value,
  // need = its 1-based rank among elements equal to T (stable-tie budget).
  unsigned prefix = 0;
  int k = NNEG;
  for (int bit = 31; bit >= 0; --bit) {
    unsigned bmask = 1u << bit;
    unsigned hmask = (bit == 31) ? 0u : ~((bmask << 1) - 1u);
    int cnt = 0;
    for (int j = 0; j < CPT; ++j) {
      int g = base + j;
      if (g < NT && (u[j] & hmask) == prefix && !(u[j] & bmask)) cnt++;
    }
    sred[t] = cnt;
    __syncthreads();
    for (int s = NEG_TPB / 2; s > 0; s >>= 1) {
      if (t < s) sred[t] += sred[t + s];
      __syncthreads();
    }
    int c0 = sred[0];
    __syncthreads();
    if (k > c0) { k -= c0; prefix |= bmask; }
  }
  unsigned T = prefix;
  int need = k;

  // Dual inclusive block scan: mask-count (-> rank) and eq-count (-> stable ties)
  int tm = 0, te = 0;
  for (int j = 0; j < CPT; ++j) { tm += m[j]; te += (u[j] == T) ? 1 : 0; }
  sred[t] = tm; sred2[t] = te;
  __syncthreads();
  for (int off = 1; off < NEG_TPB; off <<= 1) {
    int a1 = 0, a2 = 0;
    if (t >= off) { a1 = sred[t - off]; a2 = sred2[t - off]; }
    __syncthreads();
    sred[t] += a1; sred2[t] += a2;
    __syncthreads();
  }
  int runM = (t > 0) ? sred[t - 1] : 0;
  int runE = (t > 0) ? sred2[t - 1] : 0;

  float acc = 0.0f;
  for (int j = 0; j < CPT; ++j) {
    int g = base + j;
    runM += m[j];
    int e = (u[j] == T) ? 1 : 0;
    runE += e;
    bool sel = (g < NT) && ((u[j] < T) || (e && runE <= need));
    if (sel) {
      int ridx = runM - 1;
      if (ridx < 0) ridx = 0;  // JAX clamps OOB gather indices
      float c = p[((size_t)b * NT + ridx) * 25 + 4];
      float q = fminf(fmaxf(1.0f - c, EPSF), 1.0f - EPSF);
      acc -= logf(q);
    }
  }
  sfred[t] = acc;
  __syncthreads();
  for (int s = NEG_TPB / 2; s > 0; s >>= 1) {
    if (t < s) sfred[t] += sfred[t + s];
    __syncthreads();
  }
  if (t == 0) negAcc[b] = sfred[0];
}

// ---------------- Kernel 5: WMMA reduction over the 32 images ---------------
// v[b] = posAcc[b]/50 + negAcc[b];  out = mean(v).  One wave, EXEC all ones.
// A (16x4 f32): lane L<16 -> A[L][0]=v[L]; lane L>=16 -> A[L-16][2]=v[L].
// B = ones => D[M][*] = v[M] + v[M+16]; sum D rows via per-lane sum + shfl.
__global__ void k_final(const float* __restrict__ posAcc, const float* __restrict__ negAcc,
                        float* __restrict__ out) {
  int lane = threadIdx.x;  // 32 threads = 1 wave32
  float v = posAcc[lane] * (1.0f / (float)NBOX) + negAcc[lane];
#if defined(__has_builtin) && __has_builtin(__builtin_amdgcn_wmma_f32_16x16x4_f32)
  v2f a = {v, 0.0f};
  v2f bm = {1.0f, 1.0f};
  v8f c = {};
  c = __builtin_amdgcn_wmma_f32_16x16x4_f32(false, a, false, bm, (short)0, c, false, false);
#else
  // fallback: codegen-confirmed f16 WMMA (A[M][0]=v[M], A[M][8]=v[M+16])
  v16h a = {};
  a[0] = (_Float16)v;
  v16h bm;
  for (int i = 0; i < 16; ++i) bm[i] = (_Float16)1.0f;
  v8f c = {};
  c = __builtin_amdgcn_wmma_f32_16x16x32_f16(false, a, false, bm, (short)0, c, false, false);
#endif
  float s = c[0] + c[1] + c[2] + c[3] + c[4] + c[5] + c[6] + c[7];
  float total = __shfl(s, 0, 32) + __shfl(s, 16, 32);
  if (lane == 0) out[0] = total * (1.0f / (float)BIMG);
}

extern "C" void kernel_launch(void* const* d_in, const int* in_sizes, int n_in,
                              void* d_out, int out_size, void* d_ws, size_t ws_size,
                              hipStream_t stream) {
  (void)in_sizes; (void)n_in; (void)out_size; (void)ws_size;
  const float* p = (const float*)d_in[0];
  const float* boxes = (const float*)d_in[1];
  const int* labels = (const int*)d_in[2];
  float* out = (float*)d_out;

  char* ws = (char*)d_ws;
  float4* ltrb = (float4*)ws;                                  // B*NT*16 bytes
  float* mx = (float*)(ws + (size_t)BIMG * NT * 16);           // B*NT*4 bytes
  float* predAt = (float*)(ws + (size_t)BIMG * NT * 20);       // B*NT*4 bytes
  float* posAcc = (float*)(ws + (size_t)BIMG * NT * 24);       // B floats
  float* negAcc = posAcc + BIMG;                               // B floats

  k_pred_ltrb<<<dim3((BIMG * NT + 255) / 256), 256, 0, stream>>>(p, ltrb, predAt);
  k_pos<<<dim3(BIMG), 64, 0, stream>>>(p, boxes, labels, ltrb, predAt, posAcc);
  k_mx<<<dim3((NT + 255) / 256, BIMG), 256, 0, stream>>>(ltrb, predAt, boxes, mx);
  k_neg<<<dim3(BIMG), NEG_TPB, 0, stream>>>(p, mx, negAcc);
  k_final<<<1, 32, 0, stream>>>(posAcc, negAcc, out);
}
